// PF_137438954337
// MI455X (gfx1250) — matrix operand order repaired and checked by reading, not compiled, not run on hardware
//
#include <hip/hip_runtime.h>
#include <hip/hip_bf16.h>

typedef __bf16 bf16;
typedef __attribute__((ext_vector_type(16))) __bf16 v16bf;
typedef __attribute__((ext_vector_type(8)))  float  v8f;

union BV16 { v16bf v; uint4 q[2]; };

__device__ __forceinline__ bf16 f2bf(float f) {
  union { float f; unsigned u; } in; in.f = f;
  unsigned r = in.u + 0x7fffu + ((in.u >> 16) & 1u);   // round-to-nearest-even
  union { unsigned short s; bf16 b; } out; out.s = (unsigned short)(r >> 16);
  return out.b;
}

__device__ __forceinline__ float wred_sum(float v) {
#pragma unroll
  for (int m = 16; m >= 1; m >>= 1) v += __shfl_xor(v, m, 32);
  return v;
}

// ---------------- conversion kernels ----------------

// x (B,N,W) f32 -> Xb bf16 [b][t][n]
__global__ void convert_x(const float* __restrict__ x, bf16* __restrict__ Xb) {
  int idx = blockIdx.x * 256 + threadIdx.x;            // 128*64*512
  int n = idx & 511;
  int t = (idx >> 9) & 63;
  int b = idx >> 15;
  Xb[idx] = f2bf(x[((size_t)b * 512 + n) * 64 + t]);
}

// W (N,N,KS) f32 -> Wb bf16 [k][o][i]
__global__ void convert_w(const float* __restrict__ Wsrc, bf16* __restrict__ Wdst) {
  int idx = blockIdx.x * 256 + threadIdx.x;            // 3*512*512
  int i = idx & 511;
  int o = (idx >> 9) & 511;
  int k = idx >> 18;
  Wdst[idx] = f2bf(Wsrc[((size_t)o * 512 + i) * 3 + k]);
}

// Wl (W,F) f32 -> WlT bf16 [f][w]
__global__ void convert_wl(const float* __restrict__ Wl, bf16* __restrict__ WlT) {
  int idx = blockIdx.x * 256 + threadIdx.x;            // 64*64
  int w = idx & 63;
  int f = idx >> 6;
  WlT[idx] = f2bf(Wl[w * 64 + f]);
}

// ---------------- per-node scalars ----------------

__global__ void node_stats(const float* __restrict__ s_emb, const float* __restrict__ t_emb,
                           const float* __restrict__ g1_aei, const float* __restrict__ g1_aej,
                           const float* __restrict__ g3_aei, const float* __restrict__ g3_aej,
                           const float* __restrict__ Wo, const float* __restrict__ bo,
                           float* inv_s, float* inv_t, float* td1, float* sd1,
                           float* td3, float* sd3, float* cn) {
  int lane = threadIdx.x & 31;
  int n = blockIdx.x * 8 + (threadIdx.x >> 5);         // 512
  float s0 = s_emb[n * 64 + lane], s1 = s_emb[n * 64 + lane + 32];
  float t0 = t_emb[n * 64 + lane], t1 = t_emb[n * 64 + lane + 32];
  float ss = wred_sum(s0 * s0 + s1 * s1);
  float tt = wred_sum(t0 * t0 + t1 * t1);
  float a = wred_sum(t0 * g1_aei[lane] + t1 * g1_aei[lane + 32]);
  float b = wred_sum(s0 * g1_aej[lane] + s1 * g1_aej[lane + 32]);
  float c = wred_sum(t0 * g3_aei[lane] + t1 * g3_aei[lane + 32]);
  float d = wred_sum(s0 * g3_aej[lane] + s1 * g3_aej[lane + 32]);
  float e = wred_sum(s0 * Wo[128 + lane] + s1 * Wo[128 + lane + 32] +
                     t0 * Wo[192 + lane] + t1 * Wo[192 + lane + 32]);
  if (lane == 0) {
    inv_s[n] = rsqrtf(ss); inv_t[n] = rsqrtf(tt);
    td1[n] = a; sd1[n] = b; td3[n] = c; sd3[n] = d;
    cn[n] = e + bo[0];
  }
}

// ---------------- top-k neighbor build (top-26, tie by lowest index) -------

__global__ void build_srcs(const float* __restrict__ s_emb, const float* __restrict__ t_emb,
                           const float* __restrict__ inv_s, const float* __restrict__ inv_t,
                           int* __restrict__ srcs /* 512*26 */) {
  __shared__ float ti[8][64];
  int lane = threadIdx.x & 31;
  int wave = threadIdx.x >> 5;
  int i = blockIdx.x * 8 + wave;                       // 512
  float it = inv_t[i];
  ti[wave][lane]      = t_emb[i * 64 + lane] * it;
  ti[wave][lane + 32] = t_emb[i * 64 + lane + 32] * it;
  __syncthreads();

  float vals[16];
#pragma unroll
  for (int c = 0; c < 16; ++c) {
    int j = lane + 32 * c;
    const float* sj = s_emb + (size_t)j * 64;
    float acc = 0.f;
    for (int d = 0; d < 64; ++d) acc += sj[d] * ti[wave][d];
    float v = acc * inv_s[j];
    v = (v > 0.f) ? v : 0.f;                           // relu(cosine)
    vals[c] = (j == i) ? -__builtin_inff() : v;        // mask diagonal
  }
  for (int r = 0; r < 26; ++r) {
    float bv = -__builtin_inff(); int bj = 0x7fffffff;
#pragma unroll
    for (int c = 0; c < 16; ++c) {
      int j = lane + 32 * c;
      if (vals[c] > bv) { bv = vals[c]; bj = j; }      // ascending j: keep lowest on tie
    }
#pragma unroll
    for (int msk = 16; msk >= 1; msk >>= 1) {
      float ov = __shfl_xor(bv, msk, 32);
      int   oj = __shfl_xor(bj, msk, 32);
      if (ov > bv || (ov == bv && oj < bj)) { bv = ov; bj = oj; }
    }
    if (lane == 0) srcs[i * 26 + r] = bj;
    if ((bj & 31) == lane) vals[bj >> 5] = -__builtin_inff();
  }
}

// ---------------- TCN conv layer: WMMA bf16 GEMM -----------------
// Y[b,o,t] = relu( sum_k Wk[o,:]·Zin[b,:,t-(2-k)*dil] + bias[o] )  (+res,+relu)
// Zin bf16 [b][t][i]; Wb bf16 [k][o][i];
// OUT_NW=0: Zout bf16 [b][t][o] ; OUT_NW=1: Zout bf16 [b][o][t] (with residual)
// Each wave: 2 o-tiles x 4 t-tiles -> 24 WMMA per K-step vs 36 b128 loads.

template <bool OUT_NW, bool RES>
__global__ void conv_wmma(const bf16* __restrict__ Zin, const bf16* __restrict__ Wb,
                          const float* __restrict__ bias, const float* __restrict__ xres,
                          bf16* __restrict__ Zout, int dil) {
  const int lane = threadIdx.x & 31;
  const int task = blockIdx.x * 8 + (threadIdx.x >> 5); // 2048 = 16 o-pairs * 128 b
  const int op = task & 15;
  const int b  = task >> 4;
  const int m  = lane & 15;
  const int h  = lane >> 4;
  const int o0 = op * 32;

  v8f acc[2][4] = {};
#pragma unroll 4
  for (int kb = 0; kb < 512; kb += 32) {
    BV16 A[2][3];
#pragma unroll
    for (int oi = 0; oi < 2; ++oi)
#pragma unroll
      for (int k = 0; k < 3; ++k) {                     // A: 16x32 bf16, rows=o, K=i
        const bf16* pa = Wb + (size_t)k * (512 * 512) +
                         (size_t)(o0 + 16 * oi + m) * 512 + kb;
        A[oi][k].q[0] = *(const uint4*)(pa + 8 * h);    // a[0..7]  = W[o][kb+8h+j]
        A[oi][k].q[1] = *(const uint4*)(pa + 16 + 8 * h);
      }
#pragma unroll
    for (int tt = 0; tt < 4; ++tt) {
#pragma unroll
      for (int k = 0; k < 3; ++k) {
        int t = tt * 16 + m - (2 - k) * dil;            // column's shifted time
        BV16 Bv;
        if (t >= 0) {                                   // causal zero-pad
          const uint4* pb = (const uint4*)(Zin + ((size_t)b * 64 + t) * 512 + kb);
          Bv.q[0] = pb[h]; Bv.q[1] = pb[2 + h];         // b[0..7] = Zin[b][t][kb+8h+j], b[8..15] = Zin[b][t][kb+16+8h+j]
        } else {
          Bv.q[0] = make_uint4(0, 0, 0, 0); Bv.q[1] = make_uint4(0, 0, 0, 0);
        }
        acc[0][tt] = __builtin_amdgcn_wmma_f32_16x16x32_bf16(
            false, A[0][k].v, false, Bv.v, (short)0, acc[0][tt], false, false);
        acc[1][tt] = __builtin_amdgcn_wmma_f32_16x16x32_bf16(
            false, A[1][k].v, false, Bv.v, (short)0, acc[1][tt], false, false);
      }
    }
  }
#pragma unroll
  for (int oi = 0; oi < 2; ++oi) {
    const int ob = o0 + 16 * oi;
#pragma unroll
    for (int tt = 0; tt < 4; ++tt) {
      int t = tt * 16 + m;                              // output column time
      if (!OUT_NW) {
        union { uint4 q; bf16 e[8]; } ou;
#pragma unroll
        for (int v = 0; v < 8; ++v) {
          float y = acc[oi][tt][v] + bias[ob + 8 * h + v];
          ou.e[v] = f2bf(fmaxf(y, 0.f));
        }
        *(uint4*)(Zout + ((size_t)b * 64 + t) * 512 + ob + 8 * h) = ou.q;
      } else {
#pragma unroll
        for (int v = 0; v < 8; ++v) {
          int o = ob + 8 * h + v;
          float y = fmaxf(acc[oi][tt][v] + bias[o], 0.f);
          if (RES) y = fmaxf(y + xres[((size_t)b * 512 + o) * 64 + t], 0.f);
          Zout[((size_t)b * 512 + o) * 64 + t] = f2bf(y);
        }
      }
    }
  }
}

// ---------------- xw = z @ Wl  (rows=(b,n), K=w=64, cols=f=64) --------------

__global__ void xw_wmma(const bf16* __restrict__ Z3 /*[row][w]*/,
                        const bf16* __restrict__ WlT /*[f][w]*/,
                        float* __restrict__ xw /*[row][64]*/) {
  const int lane = threadIdx.x & 31;
  const int rt = blockIdx.x * 8 + (threadIdx.x >> 5);   // 4096 row tiles
  const int m = lane & 15;
  const int h = lane >> 4;
  v8f acc[4] = {};
#pragma unroll
  for (int kb = 0; kb < 64; kb += 32) {
    BV16 Av;
    const bf16* pa = Z3 + ((size_t)rt * 16 + m) * 64 + kb;
    Av.q[0] = *(const uint4*)(pa + 8 * h);
    Av.q[1] = *(const uint4*)(pa + 16 + 8 * h);
#pragma unroll
    for (int ft = 0; ft < 4; ++ft) {
      BV16 Bv;
      const uint4* pb = (const uint4*)(WlT + (size_t)(ft * 16 + m) * 64 + kb);
      Bv.q[0] = pb[h]; Bv.q[1] = pb[2 + h];
      acc[ft] = __builtin_amdgcn_wmma_f32_16x16x32_bf16(
          false, Av.v, false, Bv.v, (short)0, acc[ft], false, false);
    }
  }
#pragma unroll
  for (int ft = 0; ft < 4; ++ft)
#pragma unroll
    for (int v = 0; v < 8; ++v)
      xw[((size_t)rt * 16 + 8 * h + v) * 64 + ft * 16 + m] = acc[ft][v];
}

// ---------------- p = xw·axi, q = xw·axj per (b,n) --------------------------

__global__ void pq_kernel(const float* __restrict__ xw, const float* __restrict__ axi,
                          const float* __restrict__ axj, float* __restrict__ p,
                          float* __restrict__ q) {
  int lane = threadIdx.x & 31;
  size_t row = (size_t)blockIdx.x * 8 + (threadIdx.x >> 5);  // 65536
  float x0 = xw[row * 64 + lane], x1 = xw[row * 64 + lane + 32];
  float pp = wred_sum(x0 * axi[lane] + x1 * axi[lane + 32]);
  float qq = wred_sum(x0 * axj[lane] + x1 * axj[lane + 32]);
  if (lane == 0) { p[row] = pp; q[row] = qq; }
}

// ------------- GAT stage 2: softmax, aggregate, relu, fused head dot --------

template <int KTOP, bool ACCUM>
__global__ void gat_out(const float* __restrict__ xw, const int* __restrict__ srcs,
                        const float* __restrict__ p, const float* __restrict__ q,
                        const float* __restrict__ sd, const float* __restrict__ td,
                        const float* __restrict__ gb, const float* __restrict__ Wo, int woff,
                        const float* __restrict__ cn, float* __restrict__ out) {
  const int lane = threadIdx.x & 31;
  const int row = blockIdx.x * 8 + (threadIdx.x >> 5);  // 65536 = b*512+n
  const int b = row >> 9, n = row & 511;
  const int K = KTOP + 1;

  int src = n;
  float logit = -__builtin_inff();
  if (lane < K) {
    src = (lane < KTOP) ? srcs[n * 26 + lane] : n;      // top-k + self
    float l = p[row] + q[(size_t)b * 512 + src] + td[n] + sd[src];
    logit = (l > 0.f) ? l : 0.2f * l;                   // leaky_relu
  }
  float mx = logit;
#pragma unroll
  for (int m = 16; m >= 1; m >>= 1) mx = fmaxf(mx, __shfl_xor(mx, m, 32));
  float e = (lane < K) ? __expf(logit - mx) : 0.f;
  float s = wred_sum(e);
  float alpha = e / s;

  float a0 = 0.f, a1 = 0.f;
  for (int k = 0; k < K; ++k) {
    float ak = __shfl(alpha, k, 32);
    int   sk = __shfl(src, k, 32);
    const float* xr = xw + ((size_t)b * 512 + sk) * 64;
    a0 += ak * xr[lane];
    a1 += ak * xr[lane + 32];
  }
  a0 = fmaxf(a0 + gb[lane], 0.f);
  a1 = fmaxf(a1 + gb[lane + 32], 0.f);
  float dv = wred_sum(a0 * Wo[woff + lane] + a1 * Wo[woff + lane + 32]);
  if (lane == 0) {
    if (ACCUM) out[row] += dv;
    else       out[row] = dv + cn[n];
  }
}

// ---------------------------------------------------------------------------

extern "C" void kernel_launch(void* const* d_in, const int* in_sizes, int n_in,
                              void* d_out, int out_size, void* d_ws, size_t ws_size,
                              hipStream_t stream) {
  (void)in_sizes; (void)n_in; (void)out_size; (void)ws_size;
  const float* x      = (const float*)d_in[0];
  const float* s_emb  = (const float*)d_in[1];
  const float* t_emb  = (const float*)d_in[2];
  const float* W1 = (const float*)d_in[3];
  const float* b1 = (const float*)d_in[4];
  const float* W2 = (const float*)d_in[5];
  const float* b2 = (const float*)d_in[6];
  const float* W3 = (const float*)d_in[7];
  const float* b3 = (const float*)d_in[8];
  const float* g1_W  = (const float*)d_in[9];
  const float* g1_axi = (const float*)d_in[10];
  const float* g1_axj = (const float*)d_in[11];
  const float* g1_aei = (const float*)d_in[12];
  const float* g1_aej = (const float*)d_in[13];
  const float* g1_b   = (const float*)d_in[14];
  const float* g3_W  = (const float*)d_in[15];
  const float* g3_axi = (const float*)d_in[16];
  const float* g3_axj = (const float*)d_in[17];
  const float* g3_aei = (const float*)d_in[18];
  const float* g3_aej = (const float*)d_in[19];
  const float* g3_b   = (const float*)d_in[20];
  const float* Wo     = (const float*)d_in[21];
  const float* bo     = (const float*)d_in[22];
  float* out = (float*)d_out;

  char* ws = (char*)d_ws;
  size_t off = 0;
  auto take = [&](size_t bytes) -> char* {
    char* p = ws + off;
    off = (off + bytes + 255) & ~(size_t)255;
    return p;
  };
  bf16* bufA = (bf16*)take(8388608);     // 128*64*512 bf16 (ping)
  bf16* bufB = (bf16*)take(8388608);     // (pong)
  bf16* Wb   = (bf16*)take(4718592);     // 3 layers x [k][o][i] bf16
  bf16* WlT  = (bf16*)take(8192);        // [f][w] bf16 (reused g1/g3)
  float* xw  = (float*)take(16777216);   // [b*n][64] f32 (reused g1/g3)
  float* pA  = (float*)take(262144);
  float* qA  = (float*)take(262144);
  float* inv_s = (float*)take(2048);
  float* inv_t = (float*)take(2048);
  float* td1 = (float*)take(2048);
  float* sd1 = (float*)take(2048);
  float* td3 = (float*)take(2048);
  float* sd3 = (float*)take(2048);
  float* cnv = (float*)take(2048);
  int*   srcs = (int*)take(53248);       // 512*26

  convert_x<<<16384, 256, 0, stream>>>(x, bufA);
  convert_w<<<3072, 256, 0, stream>>>(W1, Wb);
  convert_w<<<3072, 256, 0, stream>>>(W2, Wb + 786432);
  convert_w<<<3072, 256, 0, stream>>>(W3, Wb + 1572864);
  node_stats<<<64, 256, 0, stream>>>(s_emb, t_emb, g1_aei, g1_aej, g3_aei, g3_aej,
                                     Wo, bo, inv_s, inv_t, td1, sd1, td3, sd3, cnv);
  build_srcs<<<64, 256, 0, stream>>>(s_emb, t_emb, inv_s, inv_t, srcs);

  // TCN (ping-pong): A -> B -> A -> B(z, [b][n][w] layout, fused residual)
  conv_wmma<false, false><<<256, 256, 0, stream>>>(bufA, Wb,           b1, nullptr, bufB, 1);
  conv_wmma<false, false><<<256, 256, 0, stream>>>(bufB, Wb + 786432,  b2, nullptr, bufA, 2);
  conv_wmma<true,  true ><<<256, 256, 0, stream>>>(bufA, Wb + 1572864, b3, x,       bufB, 4);

  // GAT 1 (K = 20 + self), writes out = head-dot + c_n
  convert_wl<<<16, 256, 0, stream>>>(g1_W, WlT);
  xw_wmma<<<512, 256, 0, stream>>>(bufB, WlT, xw);
  pq_kernel<<<8192, 256, 0, stream>>>(xw, g1_axi, g1_axj, pA, qA);
  gat_out<20, false><<<8192, 256, 0, stream>>>(xw, srcs, pA, qA, sd1, td1, g1_b, Wo, 0, cnv, out);

  // GAT 3 (K = 26 + self), accumulates into out
  convert_wl<<<16, 256, 0, stream>>>(g3_W, WlT);
  xw_wmma<<<512, 256, 0, stream>>>(bufB, WlT, xw);
  pq_kernel<<<8192, 256, 0, stream>>>(xw, g3_axi, g3_axj, pA, qA);
  gat_out<26, true><<<8192, 256, 0, stream>>>(xw, srcs, pA, qA, sd3, td3, g3_b, Wo, 64, cnv, out);
}
